// PerformerModel_85787676770461
// MI455X (gfx1250) — matrix operand order repaired and checked by reading, not compiled, not run on hardware
//
#include <hip/hip_runtime.h>

typedef __attribute__((ext_vector_type(16))) __bf16 v16bf;
typedef __attribute__((ext_vector_type(8)))  __bf16 v8bf;
typedef __attribute__((ext_vector_type(8)))  float  v8f;
typedef __attribute__((ext_vector_type(4)))  float  v4f;
typedef unsigned int u32;
typedef unsigned long long u64;

#define DEV __device__ __forceinline__

// ---------------- model dims ----------------
constexpr int kB = 8, kS = 8192, kIN = 64, kH = 256;
constexpr int kHEADS = 8, kDH = 32, kM = 64, kL = 4;
constexpr int kFF = 1024;
constexpr int kN  = kB * kS;        // 65536 tokens
constexpr int kBH = kB * kHEADS;    // 64 (batch,head) pairs

constexpr float kNORM  = 0.42044820762685725f;  // 32^-0.25
constexpr float kDIAG  = 0.08838834764831845f;  // 0.5 * 32^-0.5
constexpr float kRATIO = 0.125f;                // 64^-0.5
constexpr float kEPS   = 1e-4f;

// ---------------- WMMA helpers ----------------
DEV v8f wmma_bf16(v16bf a, v16bf b, v8f c) {
  return __builtin_amdgcn_wmma_f32_16x16x32_bf16(false, a, false, b, (short)0, c,
                                                 false, false);
}

DEV v16bf cat8(v8bf lo, v8bf hi) {
  return __builtin_shufflevector(lo, hi, 0, 1, 2, 3, 4, 5, 6, 7,
                                 8, 9, 10, 11, 12, 13, 14, 15);
}

// A fragment, row-major source: A(r,k) = base[r*stride + k]
// 16-bit A 16x32 layout (ISA 7.12.2): lanes 0-15 kb=0, lanes 16-31 kb=8;
// VGPR0-3 hold K=kb..kb+7, VGPR4-7 hold K=kb+16..kb+23 -> two b128 loads.
DEV v16bf frag_a_rm(const __bf16* base, int stride, int lane) {
  int r  = lane & 15;
  int kb = (lane >> 4) << 3;
  const __bf16* p = base + (size_t)r * stride + kb;
  v8bf lo = *(const v8bf*)p;
  v8bf hi = *(const v8bf*)(p + 16);
  return cat8(lo, hi);
}

// B fragment from transposed (column-major / K-contiguous) source:
// B(k,c) = baseT[(col0+c)*strideT + k]; lanes 0-15 kb=0, lanes 16-31 kb=8; K=kb..kb+7, kb+16..kb+23
DEV v16bf frag_bT(const __bf16* baseT, int strideT, int col0, int lane) {
  int c  = lane & 15;
  int kb = (lane >> 4) << 3;
  const __bf16* p = baseT + (size_t)(col0 + c) * strideT + kb;
  v8bf lo = *(const v8bf*)p;
  v8bf hi = *(const v8bf*)(p + 16);
  return cat8(lo, hi);
}

// ---------------- async copy (GLOBAL_LOAD_ASYNC_TO_LDS, ASYNCcnt) ----------
DEV unsigned lds_off32(const void* p) { return (unsigned)(size_t)p; }

DEV void async_copy16(unsigned ldsOff, u64 gaddr) {
  asm volatile("global_load_async_to_lds_b128 %0, %1, off"
               :: "v"(ldsOff), "v"(gaddr) : "memory");
}
DEV void async_wait0() { asm volatile("s_wait_asynccnt 0" ::: "memory"); }

// monotonic float<->int encoding for atomicMax on floats of either sign
DEV int   encf(float v) { int i = __float_as_int(v); return i >= 0 ? i : i ^ 0x7fffffff; }
DEV float decf(int i)   { return __int_as_float(i >= 0 ? i : i ^ 0x7fffffff); }

// ---------------- elementwise kernels ----------------
__global__ void perf_f2bf(const float* __restrict__ s, __bf16* __restrict__ d, int n) {
  int i = blockIdx.x * 256 + threadIdx.x;
  if (i < n) d[i] = (__bf16)s[i];
}

// batched transpose-convert: s[l][K][O] f32 -> d[l][O][K] bf16
__global__ void perf_f2bf_t(const float* __restrict__ s, __bf16* __restrict__ d,
                            int K, int O, int total) {
  int i = blockIdx.x * 256 + threadIdx.x;
  if (i < total) {
    int ko = K * O;
    int l = i / ko, r = i - l * ko;
    int k = r / O, o = r - k * O;
    d[(size_t)l * ko + (size_t)o * K + k] = (__bf16)s[i];
  }
}

__global__ void perf_init(int* __restrict__ kmaxEnc, float* __restrict__ ctx,
                          float* __restrict__ ksum) {
  int i = blockIdx.x * 256 + threadIdx.x;
  if (i < kBH) kmaxEnc[i] = encf(-3.0e38f);
  if (i < kBH * kM) ksum[i] = 0.0f;
  if (i < kBH * kM * kDH) ctx[i] = 0.0f;
}

// layernorm over H=256, one block per token, bf16 output
__global__ void perf_ln(const float* __restrict__ h, const float* __restrict__ g,
                        const float* __restrict__ bt, __bf16* __restrict__ out) {
  __shared__ float r1[256], r2[256];
  int n = blockIdx.x, t = threadIdx.x;
  float x = h[(size_t)n * kH + t];
  r1[t] = x; r2[t] = x * x;
  __syncthreads();
#pragma unroll
  for (int s = 128; s > 0; s >>= 1) {
    if (t < s) { r1[t] += r1[t + s]; r2[t] += r2[t + s]; }
    __syncthreads();
  }
  float mu  = r1[0] * (1.0f / 256.0f);
  float var = r2[0] * (1.0f / 256.0f) - mu * mu;
  float inv = rsqrtf(var + 1e-5f);
  out[(size_t)n * kH + t] = (__bf16)((x - mu) * inv * g[t] + bt[t]);
}

// ---------------- generic WMMA GEMM ----------------
// C[N,O] = A[N,K] * W[K,O]; A row-major bf16, WT = W transposed [O][K] bf16.
// MODE 0: outF = acc (+bias if bias)           (Q/K/V)
// MODE 1: outF = acc + bias + pos[s,col]       (embedding)
// MODE 2: resid += acc + bias                  (WO / FFN2 residual)
// MODE 3: outB = gelu(acc + bias) as bf16      (FFN1)
template <int MODE>
DEV void ep_store(int row, int col, float v, int Odim, const float* bias,
                  float* outF, __bf16* outB, const float* pos, float* resid) {
  if (bias) v += bias[col];
  if (MODE == 0) {
    outF[(size_t)row * Odim + col] = v;
  } else if (MODE == 1) {
    outF[(size_t)row * Odim + col] = v + pos[(size_t)(row & (kS - 1)) * kH + col];
  } else if (MODE == 2) {
    resid[(size_t)row * Odim + col] += v;
  } else {
    float gl = 0.5f * v * (1.0f + erff(v * 0.70710678118654752f));
    outB[(size_t)row * Odim + col] = (__bf16)gl;
  }
}

template <int MODE>
__global__ void perf_gemm(const __bf16* __restrict__ A, const __bf16* __restrict__ WT,
                          const float* __restrict__ bias, int Kdim, int Odim,
                          float* __restrict__ outF, __bf16* __restrict__ outB,
                          const float* __restrict__ pos, float* __restrict__ resid) {
  __shared__ alignas(16) __bf16 As[64 * 32];    // [row][k]   4KB
  __shared__ alignas(16) __bf16 WsT[128 * 32];  // [col][k]   8KB
  int t = threadIdx.x, lane = t & 31, wid = t >> 5;
  int rowBase = blockIdx.y * 64, colBase = blockIdx.x * 128;
  int wr = wid >> 2, wc = wid & 3;  // 2x4 wave grid, 32x32 per wave

  v8f a00 = {0, 0, 0, 0, 0, 0, 0, 0}, a01 = a00, a10 = a00, a11 = a00;

  unsigned asBase = lds_off32(As);
  unsigned wsBase = lds_off32(WsT);

  for (int kb = 0; kb < Kdim; kb += 32) {
    // stage A tile 64x32 (4KB) via one async b128 per lane
    {
      int row = t >> 2, q = t & 3;
      async_copy16(asBase + t * 16,
                   (u64)(size_t)(A + (size_t)(rowBase + row) * Kdim + kb + q * 8));
    }
    // stage WT tile 128x32 (8KB) via two async b128 per lane
#pragma unroll
    for (int p = 0; p < 2; ++p) {
      int idx = t + p * 256;
      int o = idx >> 2, q = idx & 3;
      async_copy16(wsBase + idx * 16,
                   (u64)(size_t)(WT + (size_t)(colBase + o) * Kdim + kb + q * 8));
    }
    async_wait0();
    __syncthreads();

    v16bf fa0 = frag_a_rm(As + (wr * 32) * 32, 32, lane);
    v16bf fa1 = frag_a_rm(As + (wr * 32 + 16) * 32, 32, lane);
    v16bf fb0 = frag_bT(WsT, 32, wc * 32, lane);
    v16bf fb1 = frag_bT(WsT, 32, wc * 32 + 16, lane);
    a00 = wmma_bf16(fa0, fb0, a00);
    a01 = wmma_bf16(fa0, fb1, a01);
    a10 = wmma_bf16(fa1, fb0, a10);
    a11 = wmma_bf16(fa1, fb1, a11);
    __syncthreads();
  }

  int r_in = (lane >> 4) * 8, c = lane & 15;
#pragma unroll
  for (int e = 0; e < 8; ++e) {
    int r0 = rowBase + wr * 32 + e + r_in;
    int c0 = colBase + wc * 32 + c;
    ep_store<MODE>(r0,      c0,      a00[e], Odim, bias, outF, outB, pos, resid);
    ep_store<MODE>(r0,      c0 + 16, a01[e], Odim, bias, outF, outB, pos, resid);
    ep_store<MODE>(r0 + 16, c0,      a10[e], Odim, bias, outF, outB, pos, resid);
    ep_store<MODE>(r0 + 16, c0 + 16, a11[e], Odim, bias, outF, outB, pos, resid);
  }
}

// ---------------- FAVOR+ feature kernels ----------------
DEV float dot32(const float* x, const float* p) {
  float s = 0.f;
#pragma unroll
  for (int d = 0; d < 32; ++d) s += x[d] * p[d];
  return s;
}

// pass 1 (keys): global max of xp per (b,h) via monotonic-int atomicMax
__global__ void perf_feat_max(const float* __restrict__ kmat,
                              const float* __restrict__ projL,
                              int* __restrict__ kmaxEnc) {
  __shared__ float pr[kM * kDH];
  int t = threadIdx.x;
#pragma unroll
  for (int i = t; i < kM * kDH; i += 256) pr[i] = projL[i];
  __syncthreads();
  int th = blockIdx.x * 256 + t;
  int n = th >> 3, head = th & 7;
  int b = n >> 13;
  const float* kv = kmat + (size_t)n * kH + head * kDH;
  float x[32];
#pragma unroll
  for (int d = 0; d < 32; ++d) x[d] = kv[d];
  float mx = -3.0e38f;
  for (int m = 0; m < kM; ++m) mx = fmaxf(mx, kNORM * dot32(x, pr + m * kDH));
  atomicMax(&kmaxEnc[b * kHEADS + head], encf(mx));
}

// pass 2: qp / kp = ratio * (exp(xp - diag - mx) + eps), bf16 out [bh, s, m]
__global__ void perf_feat_apply(const float* __restrict__ xmat,
                                const float* __restrict__ projL,
                                const int* __restrict__ kmaxEnc,
                                __bf16* __restrict__ outP, int isQuery) {
  __shared__ float pr[kM * kDH];
  int t = threadIdx.x;
#pragma unroll
  for (int i = t; i < kM * kDH; i += 256) pr[i] = projL[i];
  __syncthreads();
  int th = blockIdx.x * 256 + t;
  int n = th >> 3, head = th & 7;
  int b = n >> 13, s = n & (kS - 1);
  int bh = b * kHEADS + head;
  const float* xv = xmat + (size_t)n * kH + head * kDH;
  float x[32];
#pragma unroll
  for (int d = 0; d < 32; ++d) x[d] = xv[d];
  float diag = 0.f;
#pragma unroll
  for (int d = 0; d < 32; ++d) diag += x[d] * x[d];
  diag *= kDIAG;
  float mx;
  if (isQuery) {
    mx = -3.0e38f;
    for (int m = 0; m < kM; ++m) mx = fmaxf(mx, kNORM * dot32(x, pr + m * kDH));
  } else {
    mx = decf(kmaxEnc[bh]);
  }
  __bf16* dst = outP + ((size_t)bh * kS + s) * kM;
  for (int m = 0; m < kM; ++m) {
    float xp = kNORM * dot32(x, pr + m * kDH);
    dst[m] = (__bf16)(kRATIO * (expf(xp - diag - mx) + kEPS));
  }
}

// ctx[bh, m, e] += sum_n kp[bh,n,m] * v[b, n, head*32+e]  (WMMA, 16 seq splits)
// also accumulates ksum[bh, m] += sum_n kp[bh, n, m] from the staged tiles
__global__ void perf_context(const __bf16* __restrict__ kp, const float* __restrict__ v,
                             float* __restrict__ ctx, float* __restrict__ ksum) {
  __shared__ alignas(16) __bf16 skpT[64 * 32];  // [m][n]  (kp^T, K=n contiguous)
  __shared__ alignas(16) __bf16 svT[32 * 32];   // [e][n]  (v^T,  K=n contiguous)
  int t = threadIdx.x, lane = t & 31, wid = t >> 5;
  int bh = blockIdx.y, b = bh >> 3, head = bh & 7;
  int nBase = blockIdx.x * (kS / 16);
  int m0 = (wid >> 1) * 16, e0 = (wid & 1) * 16;  // 4x2 wave grid over 64x32
  v8f acc = {0, 0, 0, 0, 0, 0, 0, 0};
  float ksAcc = 0.f;
  int mrow = t >> 2, qq = t & 3;  // ksum partial ownership: (m, n-subgroup)
  for (int n0 = nBase; n0 < nBase + kS / 16; n0 += 32) {
    {  // kp tile 32n x 64m -> skpT[m][n]; one global b128 per thread
      int nn = t >> 3, m8 = (t & 7) * 8;
      v8bf val = *(const v8bf*)(kp + ((size_t)bh * kS + n0 + nn) * kM + m8);
#pragma unroll
      for (int j = 0; j < 8; ++j) skpT[(size_t)(m8 + j) * 32 + nn] = val[j];
    }
    {  // v tile 32n x 32e -> svT[e][n]; one global float4 per thread
      int nn = t >> 3, e4 = (t & 7) * 4;
      v4f val = *(const v4f*)(v + ((size_t)b * kS + n0 + nn) * kH + head * kDH + e4);
#pragma unroll
      for (int j = 0; j < 4; ++j) svT[(size_t)(e4 + j) * 32 + nn] = (__bf16)val[j];
    }
    __syncthreads();
    v16bf fa = frag_a_rm(skpT + m0 * 32, 32, lane);   // A(m, k=n)
    v16bf fb = frag_bT(svT, 32, e0, lane);            // B(k=n, e)
    acc = wmma_bf16(fa, fb, acc);
    {  // ksum partial from the staged kp^T tile (one b128 LDS read)
      v8bf ksv = *(const v8bf*)(skpT + (size_t)mrow * 32 + qq * 8);
#pragma unroll
      for (int j = 0; j < 8; ++j) ksAcc += (float)ksv[j];
    }
    __syncthreads();
  }
  int r_in = (lane >> 4) * 8, c = lane & 15;
#pragma unroll
  for (int e = 0; e < 8; ++e)
    atomicAdd(&ctx[(size_t)bh * (kM * kDH) + (m0 + e + r_in) * kDH + e0 + c], acc[e]);
  atomicAdd(&ksum[bh * kM + mrow], ksAcc);
}

// attn[token, head*32+e] = d_inv * (qp[token,:] @ ctx[bh,:,e]) as bf16
__global__ void perf_attn_out(const __bf16* __restrict__ qp, const float* __restrict__ ctx,
                              const float* __restrict__ ksumG, __bf16* __restrict__ attn) {
  __shared__ alignas(16) __bf16 sctxT[32 * 64];  // [e][m] (ctx^T, K=m contiguous)
  __shared__ float sks[64];
  __shared__ float dpart[256];
  __shared__ float dinv[128];
  int t = threadIdx.x, lane = t & 31, wid = t >> 5;
  int bh = blockIdx.y, b = bh >> 3, head = bh & 7;
  int sBase = blockIdx.x * 128;
#pragma unroll
  for (int p = 0; p < 2; ++p) {  // ctx [m][e] -> [e][m]; one float4 per thread
    int idx = t + p * 256;
    int m = idx >> 3, e4 = (idx & 7) * 4;
    v4f val = *(const v4f*)(ctx + (size_t)bh * (kM * kDH) + m * kDH + e4);
#pragma unroll
    for (int j = 0; j < 4; ++j) sctxT[(size_t)(e4 + j) * 64 + m] = (__bf16)val[j];
  }
  if (t < 64) sks[t] = ksumG[bh * kM + t];
  __syncthreads();
  {  // d_inv for the 128 rows, 2 threads per row, b128 qp loads
    int r = t >> 1, half = t & 1;
    const v8bf* q8 = (const v8bf*)(qp + ((size_t)bh * kS + sBase + r) * kM + half * 32);
    float s = 0.f;
#pragma unroll
    for (int u = 0; u < 4; ++u) {
      v8bf qv = q8[u];
#pragma unroll
      for (int j = 0; j < 8; ++j) s += (float)qv[j] * sks[half * 32 + u * 8 + j];
    }
    dpart[t] = s;
  }
  __syncthreads();
  if ((t & 1) == 0) { int r = t >> 1; dinv[r] = 1.0f / (dpart[t] + dpart[t + 1]); }
  __syncthreads();

  v8f acc0 = {0, 0, 0, 0, 0, 0, 0, 0}, acc1 = acc0;
  const __bf16* abase = qp + ((size_t)bh * kS + sBase + wid * 16) * kM;
#pragma unroll
  for (int ks = 0; ks < 2; ++ks) {  // K = 64 features
    v16bf fa  = frag_a_rm(abase + ks * 32, 64, lane);          // global b128 loads
    v16bf fb0 = frag_bT(sctxT + ks * 32, 64, 0, lane);
    v16bf fb1 = frag_bT(sctxT + ks * 32, 64, 16, lane);
    acc0 = wmma_bf16(fa, fb0, acc0);
    acc1 = wmma_bf16(fa, fb1, acc1);
  }
  int r_in = (lane >> 4) * 8, c = lane & 15;
#pragma unroll
  for (int e = 0; e < 8; ++e) {
    int r = wid * 16 + e + r_in;
    float di = dinv[r];
    size_t row = (size_t)b * kS + sBase + r;
    attn[row * kH + head * kDH + c]      = (__bf16)(acc0[e] * di);
    attn[row * kH + head * kDH + 16 + c] = (__bf16)(acc1[e] * di);
  }
}

// ---------------- pooling + classifier ----------------
__global__ void perf_pool(const float* __restrict__ h, float* __restrict__ pooled) {
  int b = blockIdx.x, t = threadIdx.x;
  float s = 0.f;
  for (int i = 0; i < kS; ++i) s += h[((size_t)b * kS + i) * kH + t];
  pooled[b * kH + t] = s * (1.0f / (float)kS);
}

__global__ void perf_fc(const float* __restrict__ pooled, const float* __restrict__ fw,
                        const float* __restrict__ fb, float* __restrict__ out) {
  int t = threadIdx.x;
  if (t < kB * 2) {
    int b = t >> 1, c = t & 1;
    float s = fb[c];
    for (int j = 0; j < kH; ++j) s += pooled[b * kH + j] * fw[j * 2 + c];
    out[b * 2 + c] = s;
  }
}

// ---------------- host launcher ----------------
extern "C" void kernel_launch(void* const* d_in, const int* in_sizes, int n_in,
                              void* d_out, int out_size, void* d_ws, size_t ws_size,
                              hipStream_t stream) {
  (void)in_sizes; (void)n_in; (void)out_size; (void)ws_size;
  const float* x     = (const float*)d_in[0];
  const float* emb_w = (const float*)d_in[1];
  const float* emb_b = (const float*)d_in[2];
  const float* pos   = (const float*)d_in[3];
  const float* ln1_g = (const float*)d_in[4];
  const float* ln1_b = (const float*)d_in[5];
  const float* wq    = (const float*)d_in[6];
  const float* wk    = (const float*)d_in[7];
  const float* wv    = (const float*)d_in[8];
  const float* wo    = (const float*)d_in[9];
  const float* bo    = (const float*)d_in[10];
  const float* ln2_g = (const float*)d_in[11];
  const float* ln2_b = (const float*)d_in[12];
  const float* w1    = (const float*)d_in[13];
  const float* b1    = (const float*)d_in[14];
  const float* w2    = (const float*)d_in[15];
  const float* b2    = (const float*)d_in[16];
  const float* proj  = (const float*)d_in[17];
  const float* fc_w  = (const float*)d_in[18];
  const float* fc_b  = (const float*)d_in[19];

  char* ws = (char*)d_ws;
  size_t off = 0;
  auto alloc = [&](size_t bytes) -> void* {
    void* p = ws + off;
    off = (off + bytes + 255) & ~(size_t)255;
    return p;
  };
  float*  h      = (float*) alloc((size_t)kN * kH * 4);
  __bf16* xn     = (__bf16*)alloc((size_t)kN * kH * 2);
  float*  qb     = (float*) alloc((size_t)kN * kH * 4);
  float*  kbuf   = (float*) alloc((size_t)kN * kH * 4);
  float*  vb     = (float*) alloc((size_t)kN * kH * 4);
  __bf16* qp     = (__bf16*)alloc((size_t)kBH * kS * kM * 2);
  __bf16* kp     = (__bf16*)alloc((size_t)kBH * kS * kM * 2);
  int*    kmax   = (int*)   alloc((size_t)kBH * 4);
  float*  ksum   = (float*) alloc((size_t)kBH * kM * 4);
  float*  ctx    = (float*) alloc((size_t)kBH * kM * kDH * 4);
  __bf16* attn   = (__bf16*)alloc((size_t)kN * kH * 2);
  __bf16* ff     = (__bf16*)alloc((size_t)kN * kFF * 2);
  float*  pooled = (float*) alloc((size_t)kB * kH * 4);
  __bf16* x_bf   = (__bf16*)alloc((size_t)kN * kIN * 2);
  __bf16* embT   = (__bf16*)alloc((size_t)kIN * kH * 2);
  __bf16* wqT    = (__bf16*)alloc((size_t)kL * kH * kH * 2);
  __bf16* wkT    = (__bf16*)alloc((size_t)kL * kH * kH * 2);
  __bf16* wvT    = (__bf16*)alloc((size_t)kL * kH * kH * 2);
  __bf16* woT    = (__bf16*)alloc((size_t)kL * kH * kH * 2);
  __bf16* w1T    = (__bf16*)alloc((size_t)kL * kH * kFF * 2);
  __bf16* w2T    = (__bf16*)alloc((size_t)kL * kFF * kH * 2);

  auto cvt = [&](const float* s, __bf16* d, size_t n) {
    perf_f2bf<<<dim3((unsigned)((n + 255) / 256)), dim3(256), 0, stream>>>(s, d, (int)n);
  };
  auto cvtT = [&](const float* s, __bf16* d, int K, int O, int total) {
    perf_f2bf_t<<<dim3((unsigned)((total + 255) / 256)), dim3(256), 0, stream>>>(
        s, d, K, O, total);
  };
  cvt(x, x_bf, (size_t)kN * kIN);
  cvtT(emb_w, embT, kIN, kH, kIN * kH);
  cvtT(wq, wqT, kH, kH, kL * kH * kH);
  cvtT(wk, wkT, kH, kH, kL * kH * kH);
  cvtT(wv, wvT, kH, kH, kL * kH * kH);
  cvtT(wo, woT, kH, kH, kL * kH * kH);
  cvtT(w1, w1T, kH, kFF, kL * kH * kFF);
  cvtT(w2, w2T, kFF, kH, kL * kFF * kH);

  dim3 blk(256);
  dim3 gH(kH / 128, kN / 64);     // O=256 GEMMs
  dim3 gF(kFF / 128, kN / 64);    // O=1024 GEMM
  dim3 gFeat((kN * kHEADS) / 256);

  // h = x @ emb_w + emb_b + pos
  perf_gemm<1><<<gH, blk, 0, stream>>>(x_bf, embT, emb_b, kIN, kH,
                                       h, nullptr, pos, nullptr);

  for (int l = 0; l < kL; ++l) {
    const float* projL = proj + (size_t)l * kM * kDH;
    // ln1 -> bf16
    perf_ln<<<dim3(kN), blk, 0, stream>>>(h, ln1_g + l * kH, ln1_b + l * kH, xn);
    // q,k,v
    perf_gemm<0><<<gH, blk, 0, stream>>>(xn, wqT + (size_t)l * kH * kH, nullptr,
                                         kH, kH, qb, nullptr, nullptr, nullptr);
    perf_gemm<0><<<gH, blk, 0, stream>>>(xn, wkT + (size_t)l * kH * kH, nullptr,
                                         kH, kH, kbuf, nullptr, nullptr, nullptr);
    perf_gemm<0><<<gH, blk, 0, stream>>>(xn, wvT + (size_t)l * kH * kH, nullptr,
                                         kH, kH, vb, nullptr, nullptr, nullptr);
    // attention state init
    perf_init<<<dim3((kBH * kM * kDH) / 256), blk, 0, stream>>>(kmax, ctx, ksum);
    // FAVOR+ features
    perf_feat_max<<<gFeat, blk, 0, stream>>>(kbuf, projL, kmax);
    perf_feat_apply<<<gFeat, blk, 0, stream>>>(qb, projL, kmax, qp, 1);
    perf_feat_apply<<<gFeat, blk, 0, stream>>>(kbuf, projL, kmax, kp, 0);
    // context = kp^T v (WMMA) + fused ksum, atomic accumulate over 16 splits
    perf_context<<<dim3(16, kBH), blk, 0, stream>>>(kp, vb, ctx, ksum);
    // out = (qp @ ctx) * d_inv  -> attn (bf16)
    perf_attn_out<<<dim3(kS / 128, kBH), blk, 0, stream>>>(qp, ctx, ksum, attn);
    // h += attn @ wo + bo
    perf_gemm<2><<<gH, blk, 0, stream>>>(attn, woT + (size_t)l * kH * kH, bo + l * kH,
                                         kH, kH, nullptr, nullptr, nullptr, h);
    // ln2 -> bf16
    perf_ln<<<dim3(kN), blk, 0, stream>>>(h, ln2_g + l * kH, ln2_b + l * kH, xn);
    // ff = gelu(xn @ w1 + b1) (bf16)
    perf_gemm<3><<<gF, blk, 0, stream>>>(xn, w1T + (size_t)l * kH * kFF, b1 + l * kFF,
                                         kH, kFF, nullptr, ff, nullptr, nullptr);
    // h += ff @ w2 + b2
    perf_gemm<2><<<gH, blk, 0, stream>>>(ff, w2T + (size_t)l * kFF * kH, b2 + l * kH,
                                         kFF, kH, nullptr, nullptr, nullptr, h);
  }

  perf_pool<<<dim3(kB), blk, 0, stream>>>(h, pooled);
  perf_fc<<<dim3(1), blk, 0, stream>>>(pooled, fc_w, fc_b, (float*)d_out);
}
